// NodeGNNwithAttentionLayers_63084479644012
// MI455X (gfx1250) — matrix operand order repaired and checked by reading, not compiled, not run on hardware
//
#include <hip/hip_runtime.h>
#include <hip/hip_bf16.h>
#include <math.h>

typedef __attribute__((ext_vector_type(16))) _Float16 v16h;
typedef __attribute__((ext_vector_type(8)))  float    v8f;

#define FDIM 64   // output feature dim of both layers (NHID == OPFEAT == 64)

// ---------------------------------------------------------------------------
// Convert W[KDIM,64] (f32, row-major) into f16 WMMA B-fragment layout:
//   frag = kb*4 + nb ; within a frag: lane l (n = l&15, kh = l>>4), half j
//   element = W[(kb*32 + kh*16 + j)*64 + nb*16 + n]
// Output: Wf16[frag*32*16 + l*16 + j]  -> each lane's fragment is 32B contiguous.
// ---------------------------------------------------------------------------
__global__ __launch_bounds__(256)
void convert_w_kernel(const float* __restrict__ W, _Float16* __restrict__ Wf16,
                      int total /* = KDIM*64 */)
{
    const int tid = blockIdx.x * blockDim.x + threadIdx.x;
    if (tid >= total) return;
    const int j    = tid & 15;
    const int l    = (tid >> 4) & 31;
    const int frag = tid >> 9;
    const int nb   = frag & 3;
    const int kb   = frag >> 2;
    const int n    = l & 15;
    const int kh   = l >> 4;
    const int k    = kb * 32 + kh * 16 + j;
    Wf16[tid] = (_Float16)W[(long)k * FDIM + nb * 16 + n];
}

// ---------------------------------------------------------------------------
// GEMM  H[N,64] = X[N,KDIM] * Wf16   (f32 X converted on the fly, f16 WMMA,
// f32 accumulate). One wave32 per 16-row tile. KDIM in {128,64}. N % 16 == 0.
// A fragment (16x32 f16, ISA 7.12.2): lane m=l&15 holds row m; kh=l>>4 selects
//   K-half: half j -> K = kb*32 + kh*16 + j (contiguous order chosen here, same as B; not the ISA table order)
// B fragment: one contiguous v16h per lane from the preconverted buffer.
// D (16x16 f32): VGPR r -> row (r + 8*kh), col n = l&15.
// ---------------------------------------------------------------------------
template <int KDIM>
__global__ __launch_bounds__(256)
void gemm_wmma_kernel(const float* __restrict__ X, const _Float16* __restrict__ Wf16,
                      float* __restrict__ H, int nrows)
{
    const int wave = (blockIdx.x * blockDim.x + threadIdx.x) >> 5;
    const int lane = threadIdx.x & 31;
    const int row0 = wave * 16;
    if (row0 >= nrows) return;

    const int m  = lane & 15;       // row within tile (A) / col within tile (D)
    const int kh = lane >> 4;       // K-half select
    const long row = (long)(row0 + m);
    const v16h* __restrict__ Bf = (const v16h*)Wf16;

    v8f acc[4];
#pragma unroll
    for (int t = 0; t < 4; ++t)
#pragma unroll
        for (int r = 0; r < 8; ++r) acc[t][r] = 0.0f;

#pragma unroll
    for (int kb = 0; kb < KDIM / 32; ++kb) {
        // ---- A fragment: 4x float4 loads, convert f32 -> f16 ----
        const float4* xr = (const float4*)(X + row * KDIM + kb * 32 + kh * 16);
        const float4 x0 = xr[0];   // K + 0..3
        const float4 x1 = xr[1];   // K + 4..7
        const float4 x2 = xr[2];   // K + 8..11
        const float4 x3 = xr[3];   // K + 12..15
        v16h afrag;
        afrag[0]  = (_Float16)x0.x; afrag[1]  = (_Float16)x0.y;
        afrag[2]  = (_Float16)x0.z; afrag[3]  = (_Float16)x0.w;
        afrag[4]  = (_Float16)x1.x; afrag[5]  = (_Float16)x1.y;
        afrag[6]  = (_Float16)x1.z; afrag[7]  = (_Float16)x1.w;
        afrag[8]  = (_Float16)x2.x; afrag[9]  = (_Float16)x2.y;
        afrag[10] = (_Float16)x2.z; afrag[11] = (_Float16)x2.w;
        afrag[12] = (_Float16)x3.x; afrag[13] = (_Float16)x3.y;
        afrag[14] = (_Float16)x3.z; afrag[15] = (_Float16)x3.w;

        // ---- 4 column tiles: contiguous 32B B-fragment loads ----
#pragma unroll
        for (int nb = 0; nb < 4; ++nb) {
            const v16h bfrag = Bf[(kb * 4 + nb) * 32 + lane];
            acc[nb] = __builtin_amdgcn_wmma_f32_16x16x32_f16(
                false, afrag, false, bfrag, (short)0, acc[nb], false, false);
        }
    }

    // ---- store D ----
#pragma unroll
    for (int nb = 0; nb < 4; ++nb) {
#pragma unroll
        for (int r = 0; r < 8; ++r) {
            const int rr = row0 + r + 8 * kh;
            if (rr < nrows)
                H[(long)rr * FDIM + nb * 16 + m] = acc[nb][r];
        }
    }
}

// ---------------------------------------------------------------------------
// Per-node attention score halves: sdst[i] = h[i]·a[0:64], ssrc[i] = h[i]·a[64:128]
// ---------------------------------------------------------------------------
__global__ __launch_bounds__(256)
void scores_kernel(const float* __restrict__ h, const float* __restrict__ a,
                   float* __restrict__ sdst, float* __restrict__ ssrc, int n)
{
    const int i = blockIdx.x * blockDim.x + threadIdx.x;
    if (i >= n) return;
    const float4* hr = (const float4*)(h + (long)i * FDIM);
    const float4* ad = (const float4*)(a);
    const float4* as = (const float4*)(a + FDIM);
    float d = 0.0f, s = 0.0f;
#pragma unroll
    for (int j = 0; j < FDIM / 4; ++j) {
        const float4 hv = hr[j], av = ad[j], bv = as[j];
        d += hv.x * av.x + hv.y * av.y + hv.z * av.z + hv.w * av.w;
        s += hv.x * bv.x + hv.y * bv.y + hv.z * bv.z + hv.w * bv.w;
    }
    sdst[i] = d;
    ssrc[i] = s;
}

// ---------------------------------------------------------------------------
// Wave-per-node: leaky-relu edge scores, softmax over the node's 16 edges,
// att-weighted gather of h[col], + bias, ReLU.
// ---------------------------------------------------------------------------
__global__ __launch_bounds__(256)
void attn_agg_kernel(const float* __restrict__ h, const int* __restrict__ col,
                     const float* __restrict__ sdst, const float* __restrict__ ssrc,
                     const float* __restrict__ bias, float* __restrict__ out, int n)
{
    const int node = (blockIdx.x * blockDim.x + threadIdx.x) >> 5;
    const int lane = threadIdx.x & 31;
    if (node >= n) return;

    int   c = 0;
    float e = -INFINITY;
    if (lane < 16) {
        c = col[(long)node * 16 + lane];
        const float v = sdst[node] + ssrc[c];
        e = (v > 0.0f) ? v : 0.2f * v;              // leaky_relu(0.2)
    }
    // softmax within the group of 16 lanes (xor offsets < 16 stay in-group)
    float mx = e;
#pragma unroll
    for (int o = 8; o >= 1; o >>= 1) mx = fmaxf(mx, __shfl_xor(mx, o, 32));
    float ex = (lane < 16) ? __expf(e - mx) : 0.0f;
    float sm = ex;
#pragma unroll
    for (int o = 8; o >= 1; o >>= 1) sm += __shfl_xor(sm, o, 32);
    const float att = ex / sm;

    float acc0 = 0.0f, acc1 = 0.0f;
#pragma unroll
    for (int k = 0; k < 16; ++k) {
        const float av = __shfl(att, k, 32);
        const int   ck = __shfl(c,   k, 32);
        const float* hr = h + (long)ck * FDIM;
        acc0 += av * hr[lane];
        acc1 += av * hr[lane + 32];
    }
    const float o0 = acc0 + bias[lane];
    const float o1 = acc1 + bias[lane + 32];
    out[(long)node * FDIM + lane]      = fmaxf(o0, 0.0f);
    out[(long)node * FDIM + lane + 32] = fmaxf(o1, 0.0f);
}

// ---------------------------------------------------------------------------
extern "C" void kernel_launch(void* const* d_in, const int* in_sizes, int n_in,
                              void* d_out, int out_size, void* d_ws, size_t ws_size,
                              hipStream_t stream)
{
    const float* x   = (const float*)d_in[0];   // [N,128]
    const int*   col = (const int*)  d_in[2];   // [N*16] (edge_col; edge_row implied)
    const float* W1  = (const float*)d_in[3];   // [128,64]
    const float* a1  = (const float*)d_in[4];   // [128]
    const float* b1  = (const float*)d_in[5];   // [64]
    const float* W2  = (const float*)d_in[6];   // [64,64]
    const float* a2  = (const float*)d_in[7];   // [128]
    const float* b2  = (const float*)d_in[8];   // [64]

    const int N = in_sizes[0] / 128;            // NFEAT = 128

    float* ws   = (float*)d_ws;
    float* h    = ws;                               // N*64 f32
    float* x1   = ws + (size_t)N * FDIM;            // N*64 f32
    float* sdst = ws + (size_t)N * FDIM * 2;        // N f32
    float* ssrc = sdst + N;                         // N f32
    _Float16* wf1 = (_Float16*)(ssrc + N);          // 128*64 f16 (16 KB)
    _Float16* wf2 = wf1 + 128 * FDIM;               // 64*64  f16 (8 KB)
    float* out  = (float*)d_out;                    // N*64 f32

    const dim3 blk(256);
    const int mtiles    = (N + 15) / 16;
    const int gemm_grid = (mtiles + 7) / 8;     // 8 waves / block
    const int node_grid = (N + 255) / 256;
    const int wave_grid = (N + 7) / 8;          // wave-per-node, 8 waves / block

    // ----- preconvert weights into WMMA B-fragment layout -----
    convert_w_kernel<<<(128 * FDIM + 255) / 256, blk, 0, stream>>>(W1, wf1, 128 * FDIM);
    convert_w_kernel<<<(64  * FDIM + 255) / 256, blk, 0, stream>>>(W2, wf2, 64  * FDIM);

    // ----- layer 1 -----
    gemm_wmma_kernel<128><<<gemm_grid, blk, 0, stream>>>(x, wf1, h, N);
    scores_kernel<<<node_grid, blk, 0, stream>>>(h, a1, sdst, ssrc, N);
    attn_agg_kernel<<<wave_grid, blk, 0, stream>>>(h, col, sdst, ssrc, b1, x1, N);

    // ----- layer 2 -----
    gemm_wmma_kernel<64><<<gemm_grid, blk, 0, stream>>>(x1, wf2, h, N);
    scores_kernel<<<node_grid, blk, 0, stream>>>(h, a2, sdst, ssrc, N);
    attn_agg_kernel<<<wave_grid, blk, 0, stream>>>(h, col, sdst, ssrc, b2, out, N);
}
